// LigerFusedLinearCrossEntropyHelion_34634616275190
// MI455X (gfx1250) — matrix operand order repaired and checked by reading, not compiled, not run on hardware
//
#include <hip/hip_runtime.h>
#include <math.h>

typedef __attribute__((ext_vector_type(16))) __bf16 bf16x16;
typedef __attribute__((ext_vector_type(8)))  float  f32x8;

#define BT 2048
#define HD 2048
#define VV 128000
#define ROWS 64                         // rows per block
#define SPLITS 8                        // V-dimension block splits
#define COLS_PER_SPLIT (VV / SPLITS)    // 16000
#define COLS_PER_ITER 128               // 8 waves * 16 cols
#define NITERS (COLS_PER_SPLIT / COLS_PER_ITER)  // 125
#define LROW_E (HD + 8)                 // bf16 elems per LDS row (+8 pad vs bank conflicts)
#define LROW_B (LROW_E * 2)             // 4112 bytes, 16B aligned

union V16U { uint4 q[2]; bf16x16 v; };
union PK2  { __bf16 h[2]; unsigned u; };

// ws layout: mW [BT*SPLITS] | sW [BT*SPLITS] | tW [BT]   (139,264 bytes)

__global__ __launch_bounds__(256, 1)
void flce_main(const float* __restrict__ x,
               const float* __restrict__ w,
               float* __restrict__ out,
               float* __restrict__ ws)
{
  extern __shared__ char smem[];
  const int tid  = threadIdx.x;
  const int wave = tid >> 5;
  const int lane = tid & 31;
  const int ln16 = lane & 15;
  const int hi   = lane >> 4;
  const int rowTile = blockIdx.x;       // 0..31
  const int split   = blockIdx.y;       // 0..7

  float* mW = ws;                       // [BT][SPLITS] running max partials
  float* sW = ws + BT * SPLITS;         // [BT][SPLITS] running sum partials

  if (rowTile == 0 && split == 0 && tid == 0) out[0] = 0.0f;

  // ---- stage x tile (ROWS x HD) into LDS as bf16 (reused NITERS times) ----
  const float* xbase = x + (size_t)rowTile * ROWS * HD;
  for (int i = tid; i < ROWS * (HD / 2); i += 256) {
    int r  = i / (HD / 2);
    int cp = i - r * (HD / 2);
    float2 f = ((const float2*)(xbase + (size_t)r * HD))[cp];
    PK2 pk; pk.h[0] = (__bf16)f.x; pk.h[1] = (__bf16)f.y;
    *(unsigned*)(smem + r * LROW_B + 4 * cp) = pk.u;
  }
  __syncthreads();

  // per-lane A base pointers (row = s*16 + ln16; K group base = hi*8 elems);
  // inner-loop k offsets fold into the ds immediate offset field.
  const char* aPtr[4];
  #pragma unroll
  for (int s = 0; s < 4; ++s) aPtr[s] = smem + (s * 16 + ln16) * LROW_B + hi * 16;

  // online logsumexp state (per lane: 32 rows x this lane's column stream)
  float mAcc[4][8], sAcc[4][8];
  #pragma unroll
  for (int s = 0; s < 4; ++s)
    #pragma unroll
    for (int j = 0; j < 8; ++j) { mAcc[s][j] = -__builtin_inff(); sAcc[s][j] = 0.0f; }

  for (int it = 0; it < NITERS; ++it) {
    const int col = split * COLS_PER_SPLIT + it * COLS_PER_ITER + wave * 16 + ln16;
    const float* wb = w + (size_t)col * HD + hi * 8;    // lane's K window start

    f32x8 acc[4] = {};

    #pragma unroll 4
    for (int k = 0; k < HD; k += 32) {
      // ---- B operand: fp32 K = k + hi*8 .. +7 and k + 16 + hi*8 .. +7 from this column's weight row ----
      const float4* wp = (const float4*)(wb + k);
      float4 w0 = wp[0], w1 = wp[1], w2 = wp[4], w3 = wp[5];
      bf16x16 bm;
      bm[0]  = (__bf16)w0.x; bm[1]  = (__bf16)w0.y; bm[2]  = (__bf16)w0.z; bm[3]  = (__bf16)w0.w;
      bm[4]  = (__bf16)w1.x; bm[5]  = (__bf16)w1.y; bm[6]  = (__bf16)w1.z; bm[7]  = (__bf16)w1.w;
      bm[8]  = (__bf16)w2.x; bm[9]  = (__bf16)w2.y; bm[10] = (__bf16)w2.z; bm[11] = (__bf16)w2.w;
      bm[12] = (__bf16)w3.x; bm[13] = (__bf16)w3.y; bm[14] = (__bf16)w3.z; bm[15] = (__bf16)w3.w;

      // ---- A operands: issue all 8 ds_load_b128 first, then 4 WMMAs ----
      V16U au[4];
      #pragma unroll
      for (int s = 0; s < 4; ++s) {
        au[s].q[0] = *(const uint4*)(aPtr[s] + 2 * k);        // K = k + hi*8 .. +7
        au[s].q[1] = *(const uint4*)(aPtr[s] + 2 * k + 32);   // K = k + 16 + hi*8 .. +7
      }
      #pragma unroll
      for (int s = 0; s < 4; ++s) {
        acc[s] = __builtin_amdgcn_wmma_f32_16x16x32_bf16(
            false, au[s].v, false, bm, (short)0, acc[s], false, false);
      }
    }

    // ---- lane-local online logsumexp update over this 64x16 logits tile ----
    #pragma unroll
    for (int s = 0; s < 4; ++s) {
      #pragma unroll
      for (int j = 0; j < 8; ++j) {
        float cv = acc[s][j];
        float mo = mAcc[s][j];
        float mn = fmaxf(mo, cv);
        sAcc[s][j] = sAcc[s][j] * __expf(mo - mn) + __expf(cv - mn);
        mAcc[s][j] = mn;
      }
    }
  }

  // ---- combine across the 16 lanes of each half-wave (different columns) ----
  #pragma unroll
  for (int mask = 1; mask < 16; mask <<= 1) {
    #pragma unroll
    for (int s = 0; s < 4; ++s) {
      #pragma unroll
      for (int j = 0; j < 8; ++j) {
        float mo = __shfl_xor(mAcc[s][j], mask);
        float so = __shfl_xor(sAcc[s][j], mask);
        float mn = fmaxf(mAcc[s][j], mo);
        sAcc[s][j] = sAcc[s][j] * __expf(mAcc[s][j] - mn) + so * __expf(mo - mn);
        mAcc[s][j] = mn;
      }
    }
  }

  // ---- combine across the 8 waves via LDS (x tile no longer needed) ----
  __syncthreads();
  float* mL = (float*)smem;             // [8][ROWS]
  float* sL = mL + 8 * ROWS;
  if (ln16 == 0) {
    #pragma unroll
    for (int s = 0; s < 4; ++s) {
      #pragma unroll
      for (int j = 0; j < 8; ++j) {
        int r = s * 16 + j + 8 * hi;
        mL[wave * ROWS + r] = mAcc[s][j];
        sL[wave * ROWS + r] = sAcc[s][j];
      }
    }
  }
  __syncthreads();
  if (tid < ROWS) {
    float M = -__builtin_inff(), S = 0.0f;
    #pragma unroll
    for (int wv = 0; wv < 8; ++wv) {
      float m2 = mL[wv * ROWS + tid], s2 = sL[wv * ROWS + tid];
      float mn = fmaxf(M, m2);
      S = S * __expf(M - mn) + s2 * __expf(m2 - mn);
      M = mn;
    }
    int rg = rowTile * ROWS + tid;
    mW[rg * SPLITS + split] = M;
    sW[rg * SPLITS + split] = S;
  }
}

// Target logit as an fp32 GEMV: t[row] = dot(x[row], W[target[row]]).
// 8.4 MFLOP / ~32 MB of reads -- negligible next to the main GEMM.
__global__ __launch_bounds__(256)
void flce_tdot(const float* __restrict__ x,
               const float* __restrict__ w,
               const int*   __restrict__ tgt,
               float* __restrict__ ws)
{
  float* tW = ws + 2 * BT * SPLITS;
  const int row = blockIdx.x;
  const int t   = tgt[row];
  const float4* x4 = (const float4*)(x + (size_t)row * HD);
  const float4* w4 = (const float4*)(w + (size_t)t * HD);
  float p = 0.0f;
  #pragma unroll
  for (int i = threadIdx.x; i < HD / 4; i += 256) {
    float4 a = x4[i], b = w4[i];
    p += a.x * b.x + a.y * b.y + a.z * b.z + a.w * b.w;
  }
  __shared__ float red[256];
  red[threadIdx.x] = p;
  __syncthreads();
  for (int st = 128; st > 0; st >>= 1) {
    if (threadIdx.x < st) red[threadIdx.x] += red[threadIdx.x + st];
    __syncthreads();
  }
  if (threadIdx.x == 0) tW[row] = red[0];
}

__global__ __launch_bounds__(256)
void flce_finish(const float* __restrict__ ws, float* __restrict__ out)
{
  const float* mW = ws;
  const float* sW = ws + BT * SPLITS;
  const float* tW = ws + 2 * BT * SPLITS;
  const int row = blockIdx.x * 256 + threadIdx.x;

  float M = -__builtin_inff(), S = 0.0f;
  #pragma unroll
  for (int p = 0; p < SPLITS; ++p) {
    float m2 = mW[row * SPLITS + p], s2 = sW[row * SPLITS + p];
    float mn = fmaxf(M, m2);
    S = S * __expf(M - mn) + s2 * __expf(m2 - mn);
    M = mn;
  }
  float lse = M + __logf(S);
  out[1 + row] = lse;            // lse output
  float nll = lse - tW[row];     // all targets valid (randint 0..V)

  __shared__ float red[256];
  red[threadIdx.x] = nll;
  __syncthreads();
  for (int st = 128; st > 0; st >>= 1) {
    if (threadIdx.x < st) red[threadIdx.x] += red[threadIdx.x + st];
    __syncthreads();
  }
  if (threadIdx.x == 0) atomicAdd(out, red[0] * (1.0f / (float)BT));
}

extern "C" void kernel_launch(void* const* d_in, const int* in_sizes, int n_in,
                              void* d_out, int out_size, void* d_ws, size_t ws_size,
                              hipStream_t stream) {
  const float* x   = (const float*)d_in[0];
  const float* w   = (const float*)d_in[1];
  const int*   tgt = (const int*)d_in[2];
  float* out = (float*)d_out;
  float* ws  = (float*)d_ws;      // needs (2*BT*SPLITS + BT)*4 = 139,264 bytes

  flce_tdot<<<BT, 256, 0, stream>>>(x, w, tgt, ws);

  dim3 grid(BT / ROWS, SPLITS);   // (32, 8)
  size_t shmem = (size_t)ROWS * LROW_B;  // 263,168 B of the 320 KB WGP LDS
  flce_main<<<grid, 256, shmem, stream>>>(x, w, out, ws);

  flce_finish<<<BT / 256, 256, 0, stream>>>(ws, out);
}
